// MHA_64811056496877
// MI455X (gfx1250) — compile-verified, matrix operand order checked by reading, not run on hardware
//
#include <hip/hip_runtime.h>
#include <hip/hip_bf16.h>

// MHA cross-attention for MI455X (gfx1250), wave32 + WMMA bf16 + async LDS copies.
// D=512, H=8, HD=64, T=1024, V=64, F=64.

using bf16 = __bf16;
typedef __attribute__((ext_vector_type(16))) __bf16 v16bf;
typedef __attribute__((ext_vector_type(8)))  __bf16 v8bf;
typedef __attribute__((ext_vector_type(4)))  __bf16 v4bf;
typedef __attribute__((ext_vector_type(8)))  float  v8f;

// ---------------------------------------------------------------------------
// CDNA5 async global->LDS copy (ASYNCcnt-tracked), via inline asm so it works
// on both toolchains. Each call copies 16 bytes per lane straight into LDS
// with no VGPR staging.
// ---------------------------------------------------------------------------
__device__ __forceinline__ unsigned lds_off(const void* p) {
    return (unsigned)(size_t)(const __attribute__((address_space(3))) void*)p;
}

__device__ __forceinline__ void async_b128(unsigned lds_byte, const bf16* g) {
    asm volatile("global_load_async_to_lds_b128 %0, %1, off"
                 :: "v"(lds_byte), "v"((unsigned long long)(size_t)g)
                 : "memory");
}
__device__ __forceinline__ void wait_async_le3() {
    asm volatile("s_wait_asynccnt 0x3" ::: "memory");
}
__device__ __forceinline__ void wait_async_0() {
    asm volatile("s_wait_asynccnt 0x0" ::: "memory");
}

// ---------------------------------------------------------------------------
// WMMA helpers.
// 16-bit A/B fragment layout (CDNA5 ISA 7.12.2): lane L<16 holds row/col L,
// K chunks {0..7, 16..23}; lane L>=16 holds row/col L-16, K chunks {8..15, 24..31}.
// Both operands of x @ W^T are K-contiguous rows, so one loader serves A and B.
// ---------------------------------------------------------------------------
__device__ __forceinline__ v16bf load_frag(const bf16* base, int stride, int lane) {
    const int row = lane & 15;
    const int klo = (lane >> 4) << 3;            // 0 or 8
    const bf16* p = base + row * stride + klo;
    v8bf lo = *(const v8bf*)(p);                  // K = klo .. klo+7
    v8bf hi = *(const v8bf*)(p + 16);             // K = 16+klo .. 16+klo+7
    return __builtin_shufflevector(lo, hi, 0, 1, 2, 3, 4, 5, 6, 7,
                                   8, 9, 10, 11, 12, 13, 14, 15);
}

__device__ __forceinline__ v8f wmma_bf16(v16bf a, v16bf b, v8f c) {
    return __builtin_amdgcn_wmma_f32_16x16x32_bf16(false, a, false, b,
                                                   (short)0, c, false, false);
}

// ---------------------------------------------------------------------------
// f32 -> bf16 conversion (4 elems / thread, 8-byte stores).
// ---------------------------------------------------------------------------
__global__ void cvt_f32_bf16(const float* __restrict__ in, bf16* __restrict__ out, int n) {
    int i = (blockIdx.x * blockDim.x + threadIdx.x) * 4;
    if (i + 3 < n) {
        float4 f = *(const float4*)(in + i);
        v4bf o;
        o[0] = (bf16)f.x; o[1] = (bf16)f.y; o[2] = (bf16)f.z; o[3] = (bf16)f.w;
        *(v4bf*)(out + i) = o;
    }
}

// ---------------------------------------------------------------------------
// C[M,N] = A[M,K] @ W[N,K]^T + bias[N]   (torch Linear semantics)
// Block tile 128x64, K step 32; 256 threads = 8 waves in 4x2; each wave 32x32.
// Double-buffered LDS fed by async global->LDS copies (3 async ops per wave
// per stage: wait_asynccnt<=3 retires current stage, next stays in flight).
// ---------------------------------------------------------------------------
template <typename OutT>
__global__ void gemm_xwT(const bf16* __restrict__ A, const bf16* __restrict__ W,
                         const float* __restrict__ bias, OutT* __restrict__ C,
                         int M, int N, int K) {
    __shared__ __align__(16) bf16 As[2][128 * 32];
    __shared__ __align__(16) bf16 Bs[2][64 * 32];

    const int tid  = threadIdx.x;
    const int lane = tid & 31;
    const int wave = tid >> 5;
    const int mw   = wave >> 1;                  // 0..3
    const int nw   = wave & 1;                   // 0..1
    const int m0   = blockIdx.y * 128;
    const int n0   = blockIdx.x * 64;

    const unsigned aBase = lds_off(&As[0][0]);
    const unsigned bBase = lds_off(&Bs[0][0]);

    // Per-thread 16B chunk coordinates (uniform instruction count per wave).
    const int kc  = (tid & 3) << 3;              // 0,8,16,24
    const int ar0 = tid >> 2;                    // rows 0..63
    const int ar1 = ar0 + 64;                    // rows 64..127
    const int br  = tid >> 2;                    // rows 0..63

    const bf16* Arow0 = A + (size_t)(m0 + ar0) * K + kc;
    const bf16* Arow1 = A + (size_t)(m0 + ar1) * K + kc;
    const bf16* Wrow  = W + (size_t)(n0 + br ) * K + kc;

    const unsigned aOffT = (unsigned)(ar0 * 32 + kc) * 2;
    const unsigned aOffB = (unsigned)(ar1 * 32 + kc) * 2;
    const unsigned bOff  = (unsigned)(br  * 32 + kc) * 2;

    auto issue = [&](int buf, int k0) {
        const unsigned aB = aBase + (unsigned)buf * (128 * 32 * 2);
        const unsigned bB = bBase + (unsigned)buf * (64 * 32 * 2);
        async_b128(aB + aOffT, Arow0 + k0);
        async_b128(aB + aOffB, Arow1 + k0);
        async_b128(bB + bOff,  Wrow  + k0);
    };

    v8f acc[2][2] = {};
    issue(0, 0);
    int buf = 0;
    for (int k0 = 0; k0 < K; k0 += 32) {
        if (k0 + 32 < K) {                        // prefetch next stage
            issue(buf ^ 1, k0 + 32);
            wait_async_le3();                     // current stage's 3 ops done
        } else {
            wait_async_0();
        }
        __syncthreads();                          // all waves' copies visible

        const bf16* as = &As[buf][0];
        const bf16* bs = &Bs[buf][0];
        v16bf af[2], bfv[2];
#pragma unroll
        for (int mi = 0; mi < 2; ++mi)
            af[mi] = load_frag(as + (mw * 32 + mi * 16) * 32, 32, lane);
#pragma unroll
        for (int ni = 0; ni < 2; ++ni)
            bfv[ni] = load_frag(bs + (nw * 32 + ni * 16) * 32, 32, lane);
#pragma unroll
        for (int mi = 0; mi < 2; ++mi)
#pragma unroll
            for (int ni = 0; ni < 2; ++ni)
                acc[mi][ni] = wmma_bf16(af[mi], bfv[ni], acc[mi][ni]);
        __syncthreads();                          // readers done before rewrite
        buf ^= 1;
    }

    // D layout: lane<16 -> N=lane, M=j; lane>=16 -> N=lane-16, M=8+j.
    const int lcol = lane & 15;
    const int mhi  = (lane >> 4) << 3;
#pragma unroll
    for (int mi = 0; mi < 2; ++mi)
#pragma unroll
        for (int ni = 0; ni < 2; ++ni) {
            int   n     = n0 + nw * 32 + ni * 16 + lcol;
            float bv    = bias[n];
            int   mbase = m0 + mw * 32 + mi * 16 + mhi;
#pragma unroll
            for (int j = 0; j < 8; ++j)
                C[(size_t)(mbase + j) * N + n] = (OutT)(acc[mi][ni][j] + bv);
        }
}

// ---------------------------------------------------------------------------
// Fused attention: one block per (video, head).  256 threads = 8 waves.
//   S[f,t] = K_h[f,:] . q[t,:] / 8     (WMMA, f32 acc -> LDS)
//   P[t,f] = softmax over f            (scalar, row-contiguous in LDS)
//   O[d,t] = Vt[d,:] . P[t,:]          (WMMA) -> attn[v,t,h*64+d] bf16
// K-tile staged via async global->LDS; V-tile needs a transpose so it is
// staged through VGPRs.
// ---------------------------------------------------------------------------
__global__ void attn_fused(const bf16* __restrict__ qb, const bf16* __restrict__ kb,
                           const bf16* __restrict__ vb, bf16* __restrict__ attn) {
    __shared__ __align__(16) bf16  Kh[64 * 64];   // [f][d]
    __shared__ __align__(16) bf16  Vt[64 * 64];   // [d][f]  (transposed V)
    __shared__ __align__(16) float Sm[64 * 64];   // [t][f]
    __shared__ __align__(16) bf16  Pm[64 * 64];   // [t][f]

    const int tid  = threadIdx.x;
    const int lane = tid & 31;
    const int wave = tid >> 5;
    const int v    = blockIdx.x >> 3;
    const int h    = blockIdx.x & 7;

    const bf16* kbase = kb + (size_t)v * 64 * 512 + h * 64;
    const bf16* vbase = vb + (size_t)v * 64 * 512 + h * 64;

    // Stage K_h via async copies (row-major, contiguous 16B chunks).
    const unsigned khBase = lds_off(&Kh[0]);
    for (int c = tid; c < 512; c += 256) {
        int f = c >> 3, dc = (c & 7) << 3;
        async_b128(khBase + (unsigned)(f * 64 + dc) * 2,
                   kbase + (size_t)f * 512 + dc);
    }
    // Stage V_h transposed (scalar gather/scatter through VGPRs).
#pragma unroll
    for (int i = 0; i < 16; ++i) {
        int e = tid * 16 + i;
        int f = e >> 6, d = e & 63;
        Vt[d * 64 + f] = vbase[(size_t)f * 512 + d];
    }
    wait_async_0();
    __syncthreads();

    const int s0 = wave * 2;    // each wave owns 2 of 16 subtiles per phase

    for (int t0 = 0; t0 < 1024; t0 += 64) {
        // ---- logits: S[f, t] ----
        for (int s = s0; s < s0 + 2; ++s) {
            int fi = s >> 2, ti = s & 3;
            v8f acc = {};
#pragma unroll
            for (int kk = 0; kk < 2; ++kk) {
                v16bf a = load_frag(Kh + fi * 16 * 64 + kk * 32, 64, lane);
                v16bf b = load_frag(qb + (size_t)(t0 + ti * 16) * 512 + h * 64 + kk * 32,
                                    512, lane);
                acc = wmma_bf16(a, b, acc);
            }
            int tl = ti * 16 + (lane & 15);
            int fb = fi * 16 + ((lane >> 4) << 3);
#pragma unroll
            for (int j = 0; j < 8; ++j)
                Sm[tl * 64 + fb + j] = acc[j] * 0.125f;   // 1/sqrt(64)
        }
        __syncthreads();

        // ---- softmax over frames (64 per column) ----
        if (tid < 64) {
            float* row = Sm + tid * 64;
            float  m = row[0];
            for (int i = 1; i < 64; ++i) m = fmaxf(m, row[i]);
            float sum = 0.f;
            for (int i = 0; i < 64; ++i) { float e = __expf(row[i] - m); row[i] = e; sum += e; }
            float inv = 1.f / sum;
            bf16* prow = Pm + tid * 64;
            for (int i = 0; i < 64; ++i) prow[i] = (bf16)(row[i] * inv);
        }
        __syncthreads();

        // ---- O[d, t] = sum_f Vt[d,f] * P[f,t] ----
        for (int s = s0; s < s0 + 2; ++s) {
            int di = s >> 2, ti = s & 3;
            v8f acc = {};
#pragma unroll
            for (int kk = 0; kk < 2; ++kk) {
                v16bf a = load_frag(Vt + di * 16 * 64 + kk * 32, 64, lane);
                v16bf b = load_frag(Pm + ti * 16 * 64 + kk * 32, 64, lane);
                acc = wmma_bf16(a, b, acc);
            }
            int t  = t0 + ti * 16 + (lane & 15);
            int db = di * 16 + ((lane >> 4) << 3);
            v8bf o;
#pragma unroll
            for (int j = 0; j < 8; ++j) o[j] = (bf16)acc[j];
            *(v8bf*)(attn + ((size_t)(v * 1024 + t)) * 512 + h * 64 + db) = o;
        }
        // Sm reuse is guarded by the post-softmax barrier; Pm reuse by the
        // pre-softmax barrier of the next iteration.
    }
}

// ---------------------------------------------------------------------------
extern "C" void kernel_launch(void* const* d_in, const int* in_sizes, int n_in,
                              void* d_out, int out_size, void* d_ws, size_t ws_size,
                              hipStream_t stream) {
    (void)in_sizes; (void)n_in; (void)out_size; (void)ws_size;
    const float* text  = (const float*)d_in[0];
    const float* video = (const float*)d_in[1];
    const float* Wq    = (const float*)d_in[2];
    const float* bq    = (const float*)d_in[3];
    const float* Wk    = (const float*)d_in[4];
    const float* bk    = (const float*)d_in[5];
    const float* Wv    = (const float*)d_in[6];
    const float* bv    = (const float*)d_in[7];
    const float* Wo    = (const float*)d_in[8];
    const float* bo    = (const float*)d_in[9];
    float* out = (float*)d_out;

    // Workspace partition (bf16 elements), ~80 MB total.
    bf16* p       = (bf16*)d_ws;
    bf16* text_b  = p; p += 1024 * 512;
    bf16* video_b = p; p += 4096 * 512;
    bf16* Wq_b    = p; p += 512 * 512;
    bf16* Wk_b    = p; p += 512 * 512;
    bf16* Wv_b    = p; p += 512 * 512;
    bf16* Wo_b    = p; p += 512 * 512;
    bf16* q_b     = p; p += 1024 * 512;
    bf16* k_b     = p; p += 4096 * 512;
    bf16* v_b     = p; p += 4096 * 512;
    bf16* attn_b  = p; p += (size_t)65536 * 512;

    // f32 -> bf16 staging.
    cvt_f32_bf16<<<(1024 * 512) / 1024, 256, 0, stream>>>(text, text_b, 1024 * 512);
    cvt_f32_bf16<<<(4096 * 512) / 1024, 256, 0, stream>>>(video, video_b, 4096 * 512);
    cvt_f32_bf16<<<(512 * 512) / 1024, 256, 0, stream>>>(Wq, Wq_b, 512 * 512);
    cvt_f32_bf16<<<(512 * 512) / 1024, 256, 0, stream>>>(Wk, Wk_b, 512 * 512);
    cvt_f32_bf16<<<(512 * 512) / 1024, 256, 0, stream>>>(Wv, Wv_b, 512 * 512);
    cvt_f32_bf16<<<(512 * 512) / 1024, 256, 0, stream>>>(Wo, Wo_b, 512 * 512);

    // Q/K/V projections (bf16 out, f32 accumulate).
    gemm_xwT<bf16><<<dim3(512 / 64, 1024 / 128), 256, 0, stream>>>(
        text_b, Wq_b, bq, q_b, 1024, 512, 512);
    gemm_xwT<bf16><<<dim3(512 / 64, 4096 / 128), 256, 0, stream>>>(
        video_b, Wk_b, bk, k_b, 4096, 512, 512);
    gemm_xwT<bf16><<<dim3(512 / 64, 4096 / 128), 256, 0, stream>>>(
        video_b, Wv_b, bv, v_b, 4096, 512, 512);

    // Fused attention: one block per (video, head).
    attn_fused<<<64 * 8, 256, 0, stream>>>(q_b, k_b, v_b, attn_b);

    // Output projection (f32 out).
    gemm_xwT<float><<<dim3(512 / 64, 65536 / 128), 256, 0, stream>>>(
        attn_b, Wo_b, bo, out, 65536, 512, 512);
}
